// SuperGlue_13297218748995
// MI455X (gfx1250) — compile-verified, matrix operand order checked by reading, not run on hardware
//
#include <hip/hip_runtime.h>
#include <math.h>

// ---------------------------------------------------------------------------
// SuperGlue scores on gfx1250 (MI455X). All GEMMs via v_wmma_f32_16x16x32_f16
// (f16 operands, f32 accumulate). One generic strided/batched WMMA GEMM kernel
// handles projections, attention scores (q^T k), messages (V P^T), MLPs, and
// the final match-score GEMM written directly into the 2049-ld Sinkhorn matrix.
// All offsets fit in 31 bits -> int stride arithmetic (v_mul_lo_u32, no mul_u64).
// Staging loads are unconditional from clamped addresses (no exec branches),
// K-overflow zeroed by cndmask; A and B both stored row-major-in-K in LDS so
// fragments load as two contiguous 16B runs (ds_load_b128).
// Workspace requirement: ~189 MB (aliased regions).
// Input flattening assumption: insertion order of setup_inputs(), nested
// dicts/lists flattened recursively in written order:
//  0 kpts0, 1 kpts1, 2 desc0, 3 desc1,
//  4-8 kenc ws[0..4], 9-13 kenc bs[0..4], 14-17 kenc gs[0..3], 18-21 betas[0..3],
//  22 proj_w, 23 proj_b, 24 merge_w, 25 merge_b, 26 mlp_w1, 27 mlp_b1,
//  28 mlp_g1, 29 mlp_be1, 30 mlp_w2, 31 mlp_b2, 32 final_w, 33 final_b, 34 bin.
// ---------------------------------------------------------------------------

typedef __attribute__((ext_vector_type(16))) _Float16 v16h;
typedef __attribute__((ext_vector_type(8)))  float    v8f;

#define BATCH   4
#define NPTS    2048
#define DIM     336
#define NHEADS  4
#define HEADDIM 84
#define NLAYERS 10
#define M1      2049          // NPTS + 1

#define BN_RSQ  0.99999500003749969f      // 1/sqrt(1+1e-5)
#define NORM_C  (-8.3177661667193446f)    // -log(4096)
#define LOGN_C  (7.6246189861593985f)     // log(2048)

struct GemmP {
  const float* A; int ars, acs, az;
  const float* B; int brs, bcs, bz;
  float*       C; int crs, cz;
  const float* bias; const float* scale; const float* shift;
  const float* R; int rrs, rcs, rz;
  int M, N, K; float alpha; int relu;
};

// Generic f32-in / f32-out GEMM, f16 WMMA core. Block = 128 threads (4 waves),
// block tile 64x64, K-step 32. Wave w covers 32x32 quadrant (2x2 WMMA tiles).
__global__ __launch_bounds__(128) void gemm_f16_wmma(GemmP p) {
  // Both operands stored as [64 rows][40 halves] (rows = m for A, n for B),
  // columns = K offset within the 32-wide tile. 40-half stride keeps the
  // 8-half fragment runs 16B aligned for ds_load_b128.
  __shared__ _Float16 As[64 * 40];
  __shared__ _Float16 Bst[64 * 40];

  const int tid  = threadIdx.x;
  const int lane = tid & 31;
  const int wave = tid >> 5;
  const int wr = wave >> 1, wc = wave & 1;
  const int lm = lane & 15, hs = lane >> 4;
  const int mBase = blockIdx.y * 64;
  const int nBase = blockIdx.x * 64;
  const int z = blockIdx.z;
  const int aoff = p.az * z;
  const int boff = p.bz * z;
  const int Kc = p.K - 1;

  v8f acc[2][2] = {};

  for (int k0 = 0; k0 < p.K; k0 += 32) {
    // ---- phase 1: unconditional global loads from clamped addresses ----
    float va[16], vb[16];
    #pragma unroll
    for (int i = 0; i < 8; ++i) {
      int pe = tid + i * 128;              // pair index for A: 64 rows x 16 pairs
      int r = pe >> 4, c2 = (pe & 15) * 2;
      int gm = mBase + r; if (gm > p.M - 1) gm = p.M - 1;     // clamp, no zeroing
      int gk0 = k0 + c2;     if (gk0 > Kc) gk0 = Kc;
      int gk1 = k0 + c2 + 1; if (gk1 > Kc) gk1 = Kc;
      int rowo = aoff + gm * p.ars;
      va[2 * i]     = p.A[rowo + gk0 * p.acs];
      va[2 * i + 1] = p.A[rowo + gk1 * p.acs];
    }
    #pragma unroll
    for (int i = 0; i < 8; ++i) {
      int pe = tid + i * 128;              // pair index for B: 64 cols x 16 k-pairs
      int n = pe & 63, kp2 = (pe >> 6) * 2;
      int gn = nBase + n; if (gn > p.N - 1) gn = p.N - 1;
      int gk0 = k0 + kp2;     if (gk0 > Kc) gk0 = Kc;
      int gk1 = k0 + kp2 + 1; if (gk1 > Kc) gk1 = Kc;
      int colo = boff + gn * p.bcs;
      vb[2 * i]     = p.B[colo + gk0 * p.brs];
      vb[2 * i + 1] = p.B[colo + gk1 * p.brs];
    }
    // prefetch next K-tile (one byte per thread per operand)
    if (k0 + 32 < p.K) {
      int pe = tid;
      int r = pe >> 4, c2 = (pe & 15) * 2;
      int gm = mBase + r; if (gm > p.M - 1) gm = p.M - 1;
      int gk = k0 + 32 + c2; if (gk > Kc) gk = Kc;
      __builtin_prefetch(&p.A[aoff + gm * p.ars + gk * p.acs], 0, 3);
      int n = pe & 63, kp2 = (pe >> 6) * 2;
      int gn = nBase + n; if (gn > p.N - 1) gn = p.N - 1;
      int gkb = k0 + 32 + kp2; if (gkb > Kc) gkb = Kc;
      __builtin_prefetch(&p.B[boff + gn * p.bcs + gkb * p.brs], 0, 3);
    }
    // ---- phase 2: convert (zero K-overflow), pack pairs, b32 LDS stores ----
    #pragma unroll
    for (int i = 0; i < 8; ++i) {
      int pe = tid + i * 128;
      int r = pe >> 4, c2 = (pe & 15) * 2;
      union { _Float16 h[2]; unsigned u; } pk;
      pk.h[0] = (k0 + c2     < p.K) ? (_Float16)va[2 * i]     : (_Float16)0.f;
      pk.h[1] = (k0 + c2 + 1 < p.K) ? (_Float16)va[2 * i + 1] : (_Float16)0.f;
      *(unsigned*)&As[r * 40 + c2] = pk.u;
    }
    #pragma unroll
    for (int i = 0; i < 8; ++i) {
      int pe = tid + i * 128;
      int n = pe & 63, kp2 = (pe >> 6) * 2;
      union { _Float16 h[2]; unsigned u; } pk;
      pk.h[0] = (k0 + kp2     < p.K) ? (_Float16)vb[2 * i]     : (_Float16)0.f;
      pk.h[1] = (k0 + kp2 + 1 < p.K) ? (_Float16)vb[2 * i + 1] : (_Float16)0.f;
      *(unsigned*)&Bst[n * 40 + kp2] = pk.u;
    }
    __syncthreads();

    // ---- fragments per ISA 16-bit layout:
    // element j of v16h <-> K = (j>>3)*16 + (lane>>4)*8 + (j&7)
    // -> two contiguous 8-half (16B) runs per fragment.
    union { v16h v; _Float16 e[16]; } af[2], bf[2];
    #pragma unroll
    for (int r = 0; r < 2; ++r) {
      int mrow = wr * 32 + r * 16 + lm;
      #pragma unroll
      for (int j = 0; j < 16; ++j) {
        int kk = ((j >> 3) << 4) + (hs << 3) + (j & 7);
        af[r].e[j] = As[mrow * 40 + kk];
      }
    }
    #pragma unroll
    for (int s = 0; s < 2; ++s) {
      int ncol = wc * 32 + s * 16 + lm;
      #pragma unroll
      for (int j = 0; j < 16; ++j) {
        int kk = ((j >> 3) << 4) + (hs << 3) + (j & 7);
        bf[s].e[j] = Bst[ncol * 40 + kk];
      }
    }
    #pragma unroll
    for (int r = 0; r < 2; ++r)
      #pragma unroll
      for (int s = 0; s < 2; ++s)
        acc[r][s] = __builtin_amdgcn_wmma_f32_16x16x32_f16(
            false, af[r].v, false, bf[s].v, (short)0, acc[r][s], false, false);
    __syncthreads();
  }

  // ---- epilogue: C/D layout row m = e + 8*(lane>>4), col n = lane&15 ----
  const int coff = p.cz * z;
  const int roff = p.rz * z;
  #pragma unroll
  for (int r = 0; r < 2; ++r) {
    #pragma unroll
    for (int s = 0; s < 2; ++s) {
      union { v8f v; float e[8]; } cc; cc.v = acc[r][s];
      #pragma unroll
      for (int t = 0; t < 8; ++t) {
        int gm = mBase + wr * 32 + r * 16 + t + hs * 8;
        int gn = nBase + wc * 32 + s * 16 + lm;
        if (gm < p.M && gn < p.N) {
          float v = cc.e[t] * p.alpha;
          if (p.R)    v += p.R[roff + gm * p.rrs + gn * p.rcs];
          if (p.bias) v += p.bias[gm];
          if (p.scale) v = v * (p.scale[gm] * BN_RSQ) + p.shift[gm];
          if (p.relu && v < 0.f) v = 0.f;
          p.C[coff + gm * p.crs + gn] = v;
        }
      }
    }
  }
}

// softmax over rows of a [rows x cols] matrix, in place, input scaled by `scale`
__global__ __launch_bounds__(256) void softmax_rows(float* d, int cols, float scale) {
  __shared__ float red[256];
  float* row = d + (long long)blockIdx.x * cols;
  int tid = threadIdx.x;
  float mx = -__builtin_inff();
  for (int j = tid; j < cols; j += 256) mx = fmaxf(mx, row[j] * scale);
  red[tid] = mx; __syncthreads();
  for (int o = 128; o > 0; o >>= 1) { if (tid < o) red[tid] = fmaxf(red[tid], red[tid + o]); __syncthreads(); }
  mx = red[0]; __syncthreads();
  float s = 0.f;
  for (int j = tid; j < cols; j += 256) s += __expf(row[j] * scale - mx);
  red[tid] = s; __syncthreads();
  for (int o = 128; o > 0; o >>= 1) { if (tid < o) red[tid] += red[tid + o]; __syncthreads(); }
  float inv = 1.f / red[0];
  for (int j = tid; j < cols; j += 256) row[j] = __expf(row[j] * scale - mx) * inv;
}

__global__ void zero_f32(float* p, int n) {
  int t = blockIdx.x * blockDim.x + threadIdx.x;
  if (t < n) p[t] = 0.f;
}

__global__ void fill_bins(float* Z0, const float* bin) {
  int t = blockIdx.x * blockDim.x + threadIdx.x;
  if (t >= BATCH * M1) return;
  int b = t / M1, r = t % M1;
  float a = *bin;
  float* Zb = Z0 + (long long)b * M1 * M1;
  Zb[(long long)r * M1 + (M1 - 1)] = a;
  Zb[(long long)(M1 - 1) * M1 + r] = a;
}

// u[b,i] = log_mu(i) - LSE_j(Z0[b,i,j] + v[b,j]);  grid (M1, BATCH)
__global__ __launch_bounds__(256) void sink_u(const float* Z0, const float* v, float* u) {
  int i = blockIdx.x, b = blockIdx.y, tid = threadIdx.x;
  const float* row = Z0 + ((long long)b * M1 + i) * M1;
  const float* vb = v + b * M1;
  __shared__ float red[256];
  float mx = -__builtin_inff();
  for (int j = tid; j < M1; j += 256) mx = fmaxf(mx, row[j] + vb[j]);
  red[tid] = mx; __syncthreads();
  for (int o = 128; o > 0; o >>= 1) { if (tid < o) red[tid] = fmaxf(red[tid], red[tid + o]); __syncthreads(); }
  mx = red[0]; __syncthreads();
  float s = 0.f;
  for (int j = tid; j < M1; j += 256) s += __expf(row[j] + vb[j] - mx);
  red[tid] = s; __syncthreads();
  for (int o = 128; o > 0; o >>= 1) { if (tid < o) red[tid] += red[tid + o]; __syncthreads(); }
  if (tid == 0) {
    float lm = (i < NPTS) ? NORM_C : (LOGN_C + NORM_C);
    u[b * M1 + i] = lm - (mx + __logf(red[0]));
  }
}

// v[b,j] = log_nu(j) - LSE_i(Z0[b,i,j] + u[b,i]);  grid (33, BATCH), 64 cols/blk
__global__ __launch_bounds__(256) void sink_v(const float* Z0, const float* u, float* v) {
  int cj = threadIdx.x & 63, rg = threadIdx.x >> 6;
  int j = blockIdx.x * 64 + cj, b = blockIdx.y;
  const float* Zb = Z0 + (long long)b * M1 * M1;
  const float* ub = u + b * M1;
  float mx = -__builtin_inff(), s = 0.f;
  if (j < M1) {
    for (int i = rg; i < M1; i += 4) {
      float val = Zb[(long long)i * M1 + j] + ub[i];
      if (val > mx) { s = s * __expf(mx - val) + 1.f; mx = val; }
      else s += __expf(val - mx);
    }
  }
  __shared__ float smx[256], ssum[256];
  smx[threadIdx.x] = mx; ssum[threadIdx.x] = s;
  __syncthreads();
  for (int off = 128; off >= 64; off >>= 1) {
    if (threadIdx.x < off) {
      float m2 = smx[threadIdx.x + off], s2 = ssum[threadIdx.x + off];
      float m0 = smx[threadIdx.x],       s0 = ssum[threadIdx.x];
      if (m2 > m0)                { ssum[threadIdx.x] = s0 * __expf(m0 - m2) + s2; smx[threadIdx.x] = m2; }
      else if (m2 > -__builtin_inff()) { ssum[threadIdx.x] = s2 * __expf(m2 - m0) + s0; }
    }
    __syncthreads();
  }
  if (rg == 0 && j < M1) {
    float ln = (j < NPTS) ? NORM_C : (LOGN_C + NORM_C);
    v[b * M1 + j] = ln - (smx[cj] + __logf(ssum[cj]));
  }
}

__global__ void final_Z(const float* Z0, const float* u, const float* v, float* out) {
  long long t = (long long)blockIdx.x * blockDim.x + threadIdx.x;
  const long long total = (long long)BATCH * M1 * M1;
  if (t >= total) return;
  int j = (int)(t % M1);
  long long r = t / M1;
  int i = (int)(r % M1);
  int b = (int)(r / M1);
  out[t] = Z0[t] + u[b * M1 + i] + v[b * M1 + j] - NORM_C;
}

__global__ __launch_bounds__(256) void row_argmax(const float* Z, float* maxv, int* idx) {
  int t = blockIdx.x;                 // b*NPTS + i
  int b = t / NPTS, i = t % NPTS;
  const float* row = Z + ((long long)b * M1 + i) * M1;
  __shared__ float bv[256]; __shared__ int bi[256];
  float best = -__builtin_inff(); int besti = 0;
  for (int j = threadIdx.x; j < NPTS; j += 256)
    if (row[j] > best) { best = row[j]; besti = j; }
  bv[threadIdx.x] = best; bi[threadIdx.x] = besti; __syncthreads();
  for (int o = 128; o > 0; o >>= 1) {
    if (threadIdx.x < o) {
      float v2 = bv[threadIdx.x + o]; int i2 = bi[threadIdx.x + o];
      if (v2 > bv[threadIdx.x] || (v2 == bv[threadIdx.x] && i2 < bi[threadIdx.x])) {
        bv[threadIdx.x] = v2; bi[threadIdx.x] = i2;
      }
    }
    __syncthreads();
  }
  if (threadIdx.x == 0) { maxv[t] = bv[0]; idx[t] = bi[0]; }
}

__global__ __launch_bounds__(256) void col_argmax(const float* Z, int* idx) {
  int t = blockIdx.x;                 // b*NPTS + j
  int b = t / NPTS, j = t % NPTS;
  const float* Zb = Z + (long long)b * M1 * M1;
  __shared__ float bv[256]; __shared__ int bi[256];
  float best = -__builtin_inff(); int besti = 0;
  for (int i = threadIdx.x; i < NPTS; i += 256) {
    float v = Zb[(long long)i * M1 + j];
    if (v > best) { best = v; besti = i; }
  }
  bv[threadIdx.x] = best; bi[threadIdx.x] = besti; __syncthreads();
  for (int o = 128; o > 0; o >>= 1) {
    if (threadIdx.x < o) {
      float v2 = bv[threadIdx.x + o]; int i2 = bi[threadIdx.x + o];
      if (v2 > bv[threadIdx.x] || (v2 == bv[threadIdx.x] && i2 < bi[threadIdx.x])) {
        bv[threadIdx.x] = v2; bi[threadIdx.x] = i2;
      }
    }
    __syncthreads();
  }
  if (threadIdx.x == 0) idx[t] = bi[0];
}

__global__ void match0_kernel(const float* max0v, const int* idx0raw, const int* idx1raw,
                              float* out_idx0, float* out_msc0) {
  int t = blockIdx.x * blockDim.x + threadIdx.x;
  if (t >= BATCH * NPTS) return;
  int b = t / NPTS, i = t % NPTS;
  int j = idx0raw[t];
  bool mutual = (idx1raw[b * NPTS + j] == i);
  float msc = mutual ? __expf(max0v[t]) : 0.f;
  bool valid = mutual && (msc > 0.2f);
  out_idx0[t] = valid ? (float)j : -1.f;
  out_msc0[t] = msc;
}

__global__ void match1_kernel(const int* idx0raw, const int* idx1raw, const float* msc0,
                              float* out_idx1, float* out_msc1) {
  int t = blockIdx.x * blockDim.x + threadIdx.x;
  if (t >= BATCH * NPTS) return;
  int b = t / NPTS, j = t % NPTS;
  int i = idx1raw[t];
  bool mutual = (idx0raw[b * NPTS + i] == j);
  float m = mutual ? msc0[b * NPTS + i] : 0.f;
  bool valid = mutual && (m > 0.2f);
  out_idx1[t] = valid ? (float)i : -1.f;
  out_msc1[t] = m;
}

// ---------------------------------------------------------------------------

static inline void gemm(hipStream_t st,
                        const float* A, int ars, int acs, int az,
                        const float* Bm, int brs, int bcs, int bz,
                        float* C, int crs, int cz,
                        int M, int N, int K, int Z,
                        const float* bias, const float* scale, const float* shift,
                        const float* R, int rrs, int rcs, int rz,
                        float alpha, int relu) {
  GemmP p{A, ars, acs, az, Bm, brs, bcs, bz, C, crs, cz,
          bias, scale, shift, R, rrs, rcs, rz, M, N, K, alpha, relu};
  dim3 g((unsigned)((N + 63) / 64), (unsigned)((M + 63) / 64), (unsigned)Z);
  gemm_f16_wmma<<<g, 128, 0, st>>>(p);
}

extern "C" void kernel_launch(void* const* d_in, const int* in_sizes, int n_in,
                              void* d_out, int out_size, void* d_ws, size_t ws_size,
                              hipStream_t stream) {
  (void)in_sizes; (void)n_in; (void)out_size; (void)ws_size;

  const float* kpts[2]   = { (const float*)d_in[0], (const float*)d_in[1] };
  const float* descin[2] = { (const float*)d_in[2], (const float*)d_in[3] };
  const float* kenc_w[5]    = { (const float*)d_in[4],  (const float*)d_in[5],
                                (const float*)d_in[6],  (const float*)d_in[7],
                                (const float*)d_in[8] };
  const float* kenc_b[5]    = { (const float*)d_in[9],  (const float*)d_in[10],
                                (const float*)d_in[11], (const float*)d_in[12],
                                (const float*)d_in[13] };
  const float* kenc_g[4]    = { (const float*)d_in[14], (const float*)d_in[15],
                                (const float*)d_in[16], (const float*)d_in[17] };
  const float* kenc_beta[4] = { (const float*)d_in[18], (const float*)d_in[19],
                                (const float*)d_in[20], (const float*)d_in[21] };
  const float* proj_w  = (const float*)d_in[22];
  const float* proj_b  = (const float*)d_in[23];
  const float* merge_w = (const float*)d_in[24];
  const float* merge_b = (const float*)d_in[25];
  const float* mlp_w1  = (const float*)d_in[26];
  const float* mlp_b1  = (const float*)d_in[27];
  const float* mlp_g1  = (const float*)d_in[28];
  const float* mlp_be1 = (const float*)d_in[29];
  const float* mlp_w2  = (const float*)d_in[30];
  const float* mlp_b2  = (const float*)d_in[31];
  const float* final_w = (const float*)d_in[32];
  const float* final_b = (const float*)d_in[33];
  const float* bin_sc  = (const float*)d_in[34];

  float* ws = (float*)d_ws;
  const long long S_DN = (long long)BATCH * DIM * NPTS;        // 2,752,512
  float* descA0 = ws;
  float* descA1 = ws + S_DN;
  float* descB0 = ws + 2 * S_DN;
  float* descB1 = ws + 3 * S_DN;
  float* regionB = ws + 4 * S_DN;           // 7*S_DN floats, multiply-aliased
  float* qbuf = regionB;
  float* kbuf = regionB + S_DN;
  float* vbuf = regionB + 2 * S_DN;
  float* msg  = regionB + 3 * S_DN;
  float* msg2 = regionB + 4 * S_DN;
  float* hbuf = regionB + 5 * S_DN;         // 2*S_DN (2D x N x B)
  float* Z0   = regionB;                    // alias (B*2049*2049 <= 7*S_DN)
  float* ping = regionB;                    // kenc scratch alias
  float* pong = regionB + (long long)1024 * BATCH * NPTS;
  float* regionC = regionB + 7 * S_DN;
  float* prob   = regionC;                  // NHEADS*NPTS*NPTS (per batch b)
  float* mdesc0 = regionC;                  // alias after attention phase
  float* mdesc1 = regionC + S_DN;
  float* smallr = regionC + (long long)NHEADS * NPTS * NPTS;
  float* ubuf  = smallr;
  float* vbuf2 = smallr + 8448;
  float* max0v = smallr + 2 * 8448;
  int*   idx0raw = (int*)(smallr + 2 * 8448 + BATCH * NPTS);
  int*   idx1raw = idx0raw + BATCH * NPTS;

  const float inv_sqrt_hd  = 1.0f / sqrtf((float)HEADDIM);
  const float inv_sqrt_dim = 1.0f / sqrtf((float)DIM);

  // ---------------- keypoint encoder (+ transposed descriptor residual) -----
  const int chans[6] = { 3, 128, 256, 512, 1024, DIM };
  for (int img = 0; img < 2; ++img) {
    float* dout = img == 0 ? descA0 : descA1;
    const float* curB = kpts[img];
    int brs = 1, bcs = 3, bz = NPTS * 3;    // (B,N,3) transposed read
    for (int i = 0; i < 4; ++i) {
      float* out = (i & 1) ? pong : ping;
      gemm(stream, kenc_w[i], chans[i], 1, 0,
           curB, brs, bcs, bz,
           out, NPTS, chans[i + 1] * NPTS,
           chans[i + 1], NPTS, chans[i], BATCH,
           kenc_b[i], kenc_g[i], kenc_beta[i],
           nullptr, 0, 0, 0, 1.0f, 1);
      curB = out; brs = NPTS; bcs = 1; bz = chans[i + 1] * NPTS;
    }
    // stage 4: plain conv + residual (descriptors transposed (B,N,D)->(B,D,N))
    gemm(stream, kenc_w[4], 1024, 1, 0,
         curB, NPTS, 1, 1024 * NPTS,
         dout, NPTS, DIM * NPTS,
         DIM, NPTS, 1024, BATCH,
         kenc_b[4], nullptr, nullptr,
         descin[img], 1, DIM, NPTS * DIM, 1.0f, 0);
  }

  // ---------------- GNN layers ---------------------------------------------
  float* cur0 = descA0; float* cur1 = descA1;
  float* nxt0 = descB0; float* nxt1 = descB1;
  for (int l = 0; l < NLAYERS; ++l) {
    const float* pw = proj_w + (long long)l * 3 * DIM * DIM;
    const float* pb = proj_b + (long long)l * 3 * DIM;
    const float* mw = merge_w + (long long)l * DIM * DIM;
    const float* mb = merge_b + (long long)l * DIM;
    const float* w1 = mlp_w1 + (long long)l * 2 * DIM * 2 * DIM;
    const float* b1 = mlp_b1 + (long long)l * 2 * DIM;
    const float* g1 = mlp_g1 + (long long)l * 2 * DIM;
    const float* be1 = mlp_be1 + (long long)l * 2 * DIM;
    const float* w2 = mlp_w2 + (long long)l * DIM * 2 * DIM;
    const float* b2 = mlp_b2 + (long long)l * DIM;

    for (int img = 0; img < 2; ++img) {
      const float* x   = (img == 0) ? cur0 : cur1;
      const float* src = (l % 2 == 1) ? ((img == 0) ? cur1 : cur0)
                                      : ((img == 0) ? cur0 : cur1);
      float* outd = (img == 0) ? nxt0 : nxt1;

      // q/k/v projections: M=D, N=NPTS, K=D, z = batch
      gemm(stream, pw, DIM, 1, 0, x, NPTS, 1, DIM * NPTS,
           qbuf, NPTS, DIM * NPTS, DIM, NPTS, DIM, BATCH,
           pb, nullptr, nullptr, nullptr, 0, 0, 0, 1.0f, 0);
      gemm(stream, pw + DIM * DIM, DIM, 1, 0, src, NPTS, 1, DIM * NPTS,
           kbuf, NPTS, DIM * NPTS, DIM, NPTS, DIM, BATCH,
           pb + DIM, nullptr, nullptr, nullptr, 0, 0, 0, 1.0f, 0);
      gemm(stream, pw + 2 * DIM * DIM, DIM, 1, 0, src, NPTS, 1, DIM * NPTS,
           vbuf, NPTS, DIM * NPTS, DIM, NPTS, DIM, BATCH,
           pb + 2 * DIM, nullptr, nullptr, nullptr, 0, 0, 0, 1.0f, 0);

      for (int b = 0; b < BATCH; ++b) {
        const float* qb = qbuf + (long long)b * DIM * NPTS;
        const float* kb = kbuf + (long long)b * DIM * NPTS;
        const float* vb = vbuf + (long long)b * DIM * NPTS;
        // scores[h][n][m] = sum_i q[(i*4+h)][n] * k[(i*4+h)][m]; z = head
        gemm(stream, qb, 1, 4 * NPTS, NPTS,
             kb, 4 * NPTS, 1, NPTS,
             prob, NPTS, NPTS * NPTS,
             NPTS, NPTS, HEADDIM, NHEADS,
             nullptr, nullptr, nullptr, nullptr, 0, 0, 0, 1.0f, 0);
        softmax_rows<<<NHEADS * NPTS, 256, 0, stream>>>(prob, NPTS, inv_sqrt_hd);
        // msg[(i*4+h)][n] = sum_m V[(i*4+h)][m] * prob[h][n][m]; z = head
        gemm(stream, vb, 4 * NPTS, 1, NPTS,
             prob, 1, NPTS, NPTS * NPTS,
             msg + (long long)b * DIM * NPTS, 4 * NPTS, NPTS,
             HEADDIM, NPTS, NPTS, NHEADS,
             nullptr, nullptr, nullptr, nullptr, 0, 0, 0, 1.0f, 0);
      }

      // merge: msg2 = merge_w @ msg + merge_b
      gemm(stream, mw, DIM, 1, 0, msg, NPTS, 1, DIM * NPTS,
           msg2, NPTS, DIM * NPTS, DIM, NPTS, DIM, BATCH,
           mb, nullptr, nullptr, nullptr, 0, 0, 0, 1.0f, 0);

      // MLP (concat-free): h = relu(bn(W1[:, :D]@x + W1[:, D:]@msg2 + b1))
      gemm(stream, w1, 2 * DIM, 1, 0, x, NPTS, 1, DIM * NPTS,
           hbuf, NPTS, 2 * DIM * NPTS, 2 * DIM, NPTS, DIM, BATCH,
           nullptr, nullptr, nullptr, nullptr, 0, 0, 0, 1.0f, 0);
      gemm(stream, w1 + DIM, 2 * DIM, 1, 0, msg2, NPTS, 1, DIM * NPTS,
           hbuf, NPTS, 2 * DIM * NPTS, 2 * DIM, NPTS, DIM, BATCH,
           b1, g1, be1, hbuf, NPTS, 1, 2 * DIM * NPTS, 1.0f, 1);
      // out = W2 @ h + b2 + x (residual)
      gemm(stream, w2, 2 * DIM, 1, 0, hbuf, NPTS, 1, 2 * DIM * NPTS,
           outd, NPTS, DIM * NPTS, DIM, NPTS, 2 * DIM, BATCH,
           b2, nullptr, nullptr, x, NPTS, 1, DIM * NPTS, 1.0f, 0);
    }
    float* t0 = cur0; cur0 = nxt0; nxt0 = t0;
    float* t1 = cur1; cur1 = nxt1; nxt1 = t1;
  }

  // ---------------- final projection + match scores into Z0 ----------------
  gemm(stream, final_w, DIM, 1, 0, cur0, NPTS, 1, DIM * NPTS,
       mdesc0, NPTS, DIM * NPTS, DIM, NPTS, DIM, BATCH,
       final_b, nullptr, nullptr, nullptr, 0, 0, 0, 1.0f, 0);
  gemm(stream, final_w, DIM, 1, 0, cur1, NPTS, 1, DIM * NPTS,
       mdesc1, NPTS, DIM * NPTS, DIM, NPTS, DIM, BATCH,
       final_b, nullptr, nullptr, nullptr, 0, 0, 0, 1.0f, 0);
  // Z0[:2048,:2048] = (mdesc0^T mdesc1) / sqrt(D), ldc = 2049
  gemm(stream, mdesc0, 1, NPTS, DIM * NPTS,
       mdesc1, NPTS, 1, DIM * NPTS,
       Z0, M1, M1 * M1,
       NPTS, NPTS, DIM, BATCH,
       nullptr, nullptr, nullptr, nullptr, 0, 0, 0, inv_sqrt_dim, 0);
  fill_bins<<<(BATCH * M1 + 255) / 256, 256, 0, stream>>>(Z0, bin_sc);

  // ---------------- Sinkhorn (100 iters, log domain) -----------------------
  zero_f32<<<(2 * 8448 + 255) / 256, 256, 0, stream>>>(ubuf, 2 * 8448);
  for (int it = 0; it < 100; ++it) {
    sink_u<<<dim3(M1, BATCH), 256, 0, stream>>>(Z0, vbuf2, ubuf);
    sink_v<<<dim3((M1 + 63) / 64, BATCH), 256, 0, stream>>>(Z0, ubuf, vbuf2);
  }

  float* outZ = (float*)d_out;
  const long long ZSZ = (long long)BATCH * M1 * M1;
  {
    long long total = ZSZ;
    final_Z<<<(unsigned)((total + 255) / 256), 256, 0, stream>>>(Z0, ubuf, vbuf2, outZ);
  }

  // ---------------- mutual matching ----------------------------------------
  row_argmax<<<BATCH * NPTS, 256, 0, stream>>>(outZ, max0v, idx0raw);
  col_argmax<<<BATCH * NPTS, 256, 0, stream>>>(outZ, idx1raw);
  const long long BN = (long long)BATCH * NPTS;
  float* out_idx0 = outZ + ZSZ;
  float* out_idx1 = outZ + ZSZ + BN;
  float* out_msc0 = outZ + ZSZ + 2 * BN;
  float* out_msc1 = outZ + ZSZ + 3 * BN;
  match0_kernel<<<(BATCH * NPTS + 255) / 256, 256, 0, stream>>>(max0v, idx0raw, idx1raw,
                                                                out_idx0, out_msc0);
  match1_kernel<<<(BATCH * NPTS + 255) / 256, 256, 0, stream>>>(idx0raw, idx1raw, out_msc0,
                                                                out_idx1, out_msc1);
}
